// HGNNP_GCNClusterNet_23192823399159
// MI455X (gfx1250) — matrix operand order repaired and checked by reading, not compiled, not run on hardware
//
#include <hip/hip_runtime.h>
#include <hip/hip_bf16.h>
#include <cstddef>

// ---------------- problem constants (match reference) ----------------
#define NN      50000
#define EE      800000
#define NUM_E   10000
#define NFEAT   256
#define NHID    128
#define NOUT    64
#define KC      10
#define NNZ     200000
#define TEMP    5.0f
#define ITER_MAX 4

#define MAX_W_ELEMS (NFEAT * NHID)   // largest packed-W panel: 32768 bf16 = 64KB

typedef __attribute__((ext_vector_type(16))) __bf16 v16bf;
typedef __attribute__((ext_vector_type(8)))  float  v8f;

// ---------------------------------------------------------------------
// Pack W (Kdim x M, f32 row-major) into bf16 WMMA B-fragment layout.
// Block (kt,nt): K rows [kt*32,+32), cols [nt*16,+16).
//   lane l: col = nb+l%16, khalf = (l>=16)?8:0
//   elems 0..7 -> K = kb+khalf+0..7, elems 8..15 -> K = kb+16+khalf+0..7
// Each lane's 16 bf16 stored contiguously (32B-aligned load in GEMM).
// ---------------------------------------------------------------------
__global__ void prep_w_bf16(const float* __restrict__ W, __bf16* __restrict__ Wp,
                            int Kdim, int M) {
    int idx = blockIdx.x * blockDim.x + threadIdx.x;
    if (idx >= Kdim * M) return;
    int e    = idx & 15;
    int lane = (idx >> 4) & 31;
    int blk  = idx >> 9;              // 512 elements per (kt,nt) block
    int NT   = M >> 4;
    int nt   = blk % NT;
    int kt   = blk / NT;
    int col  = (nt << 4) + (lane & 15);
    int kk   = (kt << 5) + ((e < 8) ? (((lane >> 4) << 3) + e) : (16 + ((lane >> 4) << 3) + e - 8));
    Wp[idx] = (__bf16)W[(size_t)kk * M + col];
}

// ---------------------------------------------------------------------
// Pack activations A (n x Kdim, f32 row-major) into bf16 WMMA A-fragment
// layout (16-bit A 16x32 per ISA):
//   lane l: row = rt*16 + l%16, khalf = (l>=16)?8:0
//   elems 0..7  -> K = kb + khalf + 0..7
//   elems 8..15 -> K = kb + 16 + khalf + 0..7
// Stored [rt][kt][lane][16] so GEMM does one 32B load per K-step.
// ---------------------------------------------------------------------
__global__ void pack_a_bf16(const float* __restrict__ A, __bf16* __restrict__ Ap,
                            int n, int Kdim) {
    int idx = blockIdx.x * blockDim.x + threadIdx.x;
    if (idx >= n * Kdim) return;
    int e      = idx & 15;
    int lane   = (idx >> 4) & 31;
    int blk    = idx >> 9;
    int ktiles = Kdim >> 5;
    int kt     = blk % ktiles;
    int rt     = blk / ktiles;
    int row    = (rt << 4) + (lane & 15);
    int khalf  = (lane >> 4) << 3;
    int kk     = (kt << 5) + ((e < 8) ? (khalf + e) : (16 + khalf + e - 8));
    Ap[idx] = (__bf16)A[(size_t)row * Kdim + kk];
}

// ---------------------------------------------------------------------
// C[n x M] = unpack(Ap) * unpack(Wp) + (bias?)
// Whole packed-W panel staged in LDS per block (cooperative copy), then
// one wave computes a 16x64 strip: 4 column tiles sharing the A fragment,
// A software-pipelined; 4x v_wmma_f32_16x16x32_bf16 per K-step with B
// sourced from ds_load.
// ---------------------------------------------------------------------
__global__ void __launch_bounds__(256)
gemm_wmma4(const __bf16* __restrict__ Ap, const __bf16* __restrict__ Wp,
           const float* __restrict__ bias, float* __restrict__ C,
           int n, int Kdim, int M) {
    __shared__ __align__(32) __bf16 sW[MAX_W_ELEMS];
    // cooperative stage of the entire packed-W panel into LDS (16B chunks)
    {
        const uint4* s4 = (const uint4*)Wp;
        uint4*       d4 = (uint4*)sW;
        const int n4 = (Kdim * M) >> 3;          // 8 bf16 per uint4
        for (int i = threadIdx.x; i < n4; i += blockDim.x) d4[i] = s4[i];
    }
    __syncthreads();

    const int lane = threadIdx.x & 31;
    const int wave = threadIdx.x >> 5;
    const int NT   = M >> 4;          // 16-col tiles
    const int WT   = NT >> 2;         // wave strips per row-tile (4 col tiles each)
    const int wt   = blockIdx.x * 8 + wave;
    const int numWT = (n >> 4) * WT;
    if (wt >= numWT) return;          // after barrier; wave-uniform
    const int ct = (wt % WT) << 2;    // first column tile of this strip
    const int rt = wt / WT;
    const int ktiles = Kdim >> 5;

    v8f acc[4] = {};
    const __bf16* apBase = Ap + ((size_t)rt * ktiles * 32 + lane) * 16;
    v16bf a = *(const v16bf*)apBase;               // prefetch kt = 0
    for (int kt = 0; kt < ktiles; ++kt) {
        v16bf a_cur = a;
        if (kt + 1 < ktiles)                        // uniform branch
            a = *(const v16bf*)(apBase + (size_t)(kt + 1) * 512);
        const __bf16* bp = sW + ((kt * NT + ct) * 32 + lane) * 16;
        v16bf b0 = *(const v16bf*)(bp);
        v16bf b1 = *(const v16bf*)(bp + 512);
        v16bf b2 = *(const v16bf*)(bp + 1024);
        v16bf b3 = *(const v16bf*)(bp + 1536);
        acc[0] = __builtin_amdgcn_wmma_f32_16x16x32_bf16(false, a_cur, false, b0, (short)0, acc[0], false, false);
        acc[1] = __builtin_amdgcn_wmma_f32_16x16x32_bf16(false, a_cur, false, b1, (short)0, acc[1], false, false);
        acc[2] = __builtin_amdgcn_wmma_f32_16x16x32_bf16(false, a_cur, false, b2, (short)0, acc[2], false, false);
        acc[3] = __builtin_amdgcn_wmma_f32_16x16x32_bf16(false, a_cur, false, b3, (short)0, acc[3], false, false);
    }
    // C/D layout: VGPR r, lane l -> row = rt*16 + r + 8*(l/16), col = nt*16 + l%16
    const int rbase = (rt << 4) + ((lane >> 4) << 3);
    const int lcol  = lane & 15;
#pragma unroll
    for (int t = 0; t < 4; ++t) {
        const int cc = ((ct + t) << 4) + lcol;
        const float bv = bias ? bias[cc] : 0.0f;
#pragma unroll
        for (int r = 0; r < 8; ++r)
            C[(size_t)(rbase + r) * M + cc] = acc[t][r] + bv;
    }
}

// ---------------------------- small helpers ---------------------------
__global__ void fill_kernel(float* __restrict__ p, float v, int n) {
    int i = blockIdx.x * blockDim.x + threadIdx.x;
    if (i < n) p[i] = v;
}

__global__ void count_kernel(const int* __restrict__ idx, float* __restrict__ cnt, int n) {
    int i = blockIdx.x * blockDim.x + threadIdx.x;
    if (i < n) atomicAdd(&cnt[idx[i]], 1.0f);
}

__global__ void norm_kernel(const int* __restrict__ src, const int* __restrict__ dst,
                            const float* __restrict__ deg, float* __restrict__ nrm, int e) {
    int i = blockIdx.x * blockDim.x + threadIdx.x;
    if (i < e) nrm[i] = rsqrtf(deg[src[i]]) * rsqrtf(deg[dst[i]]);
}

// out[didx[j]] += (w ? w[j] : 1) * in[sidx[j]], 8 features per thread
__global__ void __launch_bounds__(256)
scatter_kernel(const float* __restrict__ in, const int* __restrict__ sidx,
               const int* __restrict__ didx, const float* __restrict__ w,
               float* __restrict__ out, int nnz, int F) {
    int t = blockIdx.x * blockDim.x + threadIdx.x;
    int groups = F >> 3;
    if (t >= nnz * groups) return;
    int j = t / groups;
    int g = (t % groups) << 3;
    int s = sidx[j], d = didx[j];
    float ww = w ? w[j] : 1.0f;
    const float* hp = in + (size_t)s * F + g;
    float*       op = out + (size_t)d * F + g;
    __builtin_prefetch(hp, 0, 1);
#pragma unroll
    for (int i = 0; i < 8; ++i) atomicAdd(op + i, hp[i] * ww);
}

// GCN epilogue: out = acc + h/deg + b, optional relu (in place on acc)
__global__ void finalize_gcn(float* __restrict__ acc, const float* __restrict__ h,
                             const float* __restrict__ deg, const float* __restrict__ b,
                             int n, int F, int relu) {
    int t = blockIdx.x * blockDim.x + threadIdx.x;
    if (t >= n * F) return;
    int row = t / F, f = t % F;
    float v = acc[t] + h[t] / deg[row] + b[f];
    acc[t] = relu ? fmaxf(v, 0.0f) : v;
}

// x[row,f] /= max(cnt[row],1), optional relu
__global__ void div_rows(float* __restrict__ x, const float* __restrict__ cnt,
                         int n, int F, int relu) {
    int t = blockIdx.x * blockDim.x + threadIdx.x;
    if (t >= n * F) return;
    int row = t / F;
    float v = x[t] / fmaxf(cnt[row], 1.0f);
    x[t] = relu ? fmaxf(v, 0.0f) : v;
}

__global__ void combine_kernel(const float* __restrict__ a, const float* __restrict__ b,
                               float* __restrict__ out, int n) {
    int i = blockIdx.x * blockDim.x + threadIdx.x;
    if (i < n) out[i] = 0.5f * (a[i] + b[i]);
}

__global__ void rownorm_kernel(const float* __restrict__ emb, float* __restrict__ data) {
    int row = blockIdx.x * blockDim.x + threadIdx.x;
    if (row >= NN) return;
    const float* p = emb + (size_t)row * NOUT;
    float s = 0.0f;
#pragma unroll
    for (int f = 0; f < NOUT; ++f) { float v = p[f]; s += v * v; }
    float inv = rsqrtf(s);
    float* q = data + (size_t)row * NOUT;
#pragma unroll
    for (int f = 0; f < NOUT; ++f) q[f] = p[f] * inv;
}

__global__ void init_mu_kernel(const float* __restrict__ data, float* __restrict__ mu) {
    int t = blockIdx.x * blockDim.x + threadIdx.x;
    if (t >= KC * NOUT) return;
    int k = t / NOUT, f = t % NOUT;
    mu[t] = data[(size_t)(k * (NN / KC)) * NOUT + f];
}

__global__ void gated_zero(float* __restrict__ p, int n, const int* __restrict__ ni, int it) {
    if (it >= *ni) return;
    int i = blockIdx.x * blockDim.x + threadIdx.x;
    if (i < n) p[i] = 0.0f;
}

// one soft-kmeans E-step + accumulation; block-local LDS reduction
__global__ void __launch_bounds__(256)
kmeans_accum(const float* __restrict__ data, const float* __restrict__ mu,
             float* __restrict__ cr, float* __restrict__ cs,
             const int* __restrict__ ni, int it) {
    if (it >= *ni) return;
    __shared__ float s_mu[KC * NOUT];
    __shared__ float s_cs[KC * NOUT];
    __shared__ float s_cr[KC];
    for (int i = threadIdx.x; i < KC * NOUT; i += blockDim.x) { s_mu[i] = mu[i]; s_cs[i] = 0.0f; }
    if (threadIdx.x < KC) s_cr[threadIdx.x] = 0.0f;
    __syncthreads();
    int row = blockIdx.x * blockDim.x + threadIdx.x;
    if (row < NN) {
        const float* dp = data + (size_t)row * NOUT;
        float d[NOUT];
#pragma unroll
        for (int f = 0; f < NOUT; ++f) d[f] = dp[f];
        float lg[KC]; float mx = -3.0e38f;
#pragma unroll
        for (int k = 0; k < KC; ++k) {
            float s = 0.0f;
#pragma unroll
            for (int f = 0; f < NOUT; ++f) s += d[f] * s_mu[k * NOUT + f];
            lg[k] = TEMP * s; mx = fmaxf(mx, lg[k]);
        }
        float se = 0.0f; float r[KC];
#pragma unroll
        for (int k = 0; k < KC; ++k) { r[k] = __expf(lg[k] - mx); se += r[k]; }
        float inv = 1.0f / se;
#pragma unroll
        for (int k = 0; k < KC; ++k) {
            r[k] *= inv;
            atomicAdd(&s_cr[k], r[k]);
#pragma unroll
            for (int f = 0; f < NOUT; ++f) atomicAdd(&s_cs[k * NOUT + f], r[k] * d[f]);
        }
    }
    __syncthreads();
    for (int i = threadIdx.x; i < KC * NOUT; i += blockDim.x) atomicAdd(&cs[i], s_cs[i]);
    if (threadIdx.x < KC) atomicAdd(&cr[threadIdx.x], s_cr[threadIdx.x]);
}

__global__ void mu_update(float* __restrict__ mu, const float* __restrict__ cs,
                          const float* __restrict__ cr, const int* __restrict__ ni, int it) {
    if (it >= *ni) return;
    int t = blockIdx.x * blockDim.x + threadIdx.x;
    if (t >= KC * NOUT) return;
    mu[t] = cs[t] / cr[t / NOUT];
}

__global__ void copy_kernel(const float* __restrict__ src, float* __restrict__ dst, int n) {
    int i = blockIdx.x * blockDim.x + threadIdx.x;
    if (i < n) dst[i] = src[i];
}

__global__ void __launch_bounds__(256)
kmeans_final(const float* __restrict__ data, const float* __restrict__ mu,
             float* __restrict__ r_out, float* __restrict__ dist_out) {
    __shared__ float s_mu[KC * NOUT];
    for (int i = threadIdx.x; i < KC * NOUT; i += blockDim.x) s_mu[i] = mu[i];
    __syncthreads();
    int row = blockIdx.x * blockDim.x + threadIdx.x;
    if (row >= NN) return;
    const float* dp = data + (size_t)row * NOUT;
    float d[NOUT];
#pragma unroll
    for (int f = 0; f < NOUT; ++f) d[f] = dp[f];
    float lg[KC]; float mx = -3.0e38f;
#pragma unroll
    for (int k = 0; k < KC; ++k) {
        float s = 0.0f;
#pragma unroll
        for (int f = 0; f < NOUT; ++f) s += d[f] * s_mu[k * NOUT + f];
        dist_out[(size_t)row * KC + k] = s;
        lg[k] = TEMP * s; mx = fmaxf(mx, lg[k]);
    }
    float se = 0.0f; float r[KC];
#pragma unroll
    for (int k = 0; k < KC; ++k) { r[k] = __expf(lg[k] - mx); se += r[k]; }
    float inv = 1.0f / se;
#pragma unroll
    for (int k = 0; k < KC; ++k) r_out[(size_t)row * KC + k] = r[k] * inv;
}

// ---------------------------------------------------------------------
extern "C" void kernel_launch(void* const* d_in, const int* in_sizes, int n_in,
                              void* d_out, int out_size, void* d_ws, size_t ws_size,
                              hipStream_t stream) {
    (void)in_sizes; (void)n_in; (void)out_size; (void)ws_size;
    const float* x    = (const float*)d_in[0];
    const int*   eidx = (const int*)d_in[1];
    const int*   src  = eidx;
    const int*   dst  = eidx + EE;
    const int*   hv   = (const int*)d_in[2];
    const int*   he   = (const int*)d_in[3];
    const float* gW1 = (const float*)d_in[4];  const float* gb1 = (const float*)d_in[5];
    const float* gW2 = (const float*)d_in[6];  const float* gb2 = (const float*)d_in[7];
    const float* hW1 = (const float*)d_in[8];  const float* hb1 = (const float*)d_in[9];
    const float* hW2 = (const float*)d_in[10]; const float* hb2 = (const float*)d_in[11];
    const int*   ni  = (const int*)d_in[12];

    float* out      = (float*)d_out;
    float* out_mu   = out;                                 // [K*NOUT]
    float* out_r    = out + KC * NOUT;                     // [N*K]
    float* out_emb  = out + KC * NOUT + (size_t)NN * KC;   // [N*NOUT]
    float* out_dist = out_emb + (size_t)NN * NOUT;         // [N*K]

    // ---- workspace carve (bump allocator, 256B aligned) ----
    char* wsp = (char*)d_ws;
    auto carve = [&](size_t bytes) -> char* {
        char* p = wsp; wsp += (bytes + 255) & ~(size_t)255; return p;
    };
    __bf16* Wp_g1 = (__bf16*)carve((size_t)NFEAT * NHID * 2);
    __bf16* Wp_g2 = (__bf16*)carve((size_t)NHID * NOUT * 2);
    __bf16* Wp_h1 = (__bf16*)carve((size_t)NFEAT * NHID * 2);
    __bf16* Wp_h2 = (__bf16*)carve((size_t)NHID * NOUT * 2);
    __bf16* Ap_x  = (__bf16*)carve((size_t)NN * NFEAT * 2); // packed x (layer-1 GEMMs)
    __bf16* Ap_h  = (__bf16*)carve((size_t)NN * NHID * 2);  // packed x1 / x3
    float* BUF1  = (float*)carve((size_t)NN * NHID * 4);    // h1 / h3 / x4-acc
    float* BUF2  = (float*)carve((size_t)NN * NHID * 4);    // gcn acc1->x1 / v-acc->x3
    float* SBUF1 = (float*)carve((size_t)NN * NOUT * 4);    // h2 / h4
    float* SBUF2 = (float*)carve((size_t)NN * NOUT * 4);    // gcn acc2->x2
    float* e_acc = (float*)carve((size_t)NUM_E * NHID * 4);
    float* nrm   = (float*)carve((size_t)EE * 4);
    float* deg   = (float*)carve((size_t)NN * 4);
    float* e_deg = (float*)carve((size_t)NUM_E * 4);
    float* v_deg = (float*)carve((size_t)NN * 4);
    float* data  = (float*)carve((size_t)NN * NOUT * 4);
    float* mu    = (float*)carve((size_t)KC * NOUT * 4);
    float* cr    = (float*)carve((size_t)KC * 4);
    float* cs    = (float*)carve((size_t)KC * NOUT * 4);

    auto cdiv = [](int a, int b) { return (a + b - 1) / b; };
    const int T = 256;

    // ---- pack weights to bf16 WMMA B-fragments ----
    prep_w_bf16<<<cdiv(NFEAT * NHID, T), T, 0, stream>>>(gW1, Wp_g1, NFEAT, NHID);
    prep_w_bf16<<<cdiv(NHID * NOUT, T), T, 0, stream>>>(gW2, Wp_g2, NHID, NOUT);
    prep_w_bf16<<<cdiv(NFEAT * NHID, T), T, 0, stream>>>(hW1, Wp_h1, NFEAT, NHID);
    prep_w_bf16<<<cdiv(NHID * NOUT, T), T, 0, stream>>>(hW2, Wp_h2, NHID, NOUT);
    // ---- pack x to bf16 WMMA A-fragments (shared by GCN & HGNNP layer 1) ----
    pack_a_bf16<<<cdiv(NN * NFEAT, T), T, 0, stream>>>(x, Ap_x, NN, NFEAT);

    // ---- GCN degrees + edge norm ----
    fill_kernel<<<cdiv(NN, T), T, 0, stream>>>(deg, 1.0f, NN);   // self-loop
    count_kernel<<<cdiv(EE, T), T, 0, stream>>>(dst, deg, EE);
    norm_kernel<<<cdiv(EE, T), T, 0, stream>>>(src, dst, deg, nrm, EE);

    const int waves1 = (NN / 16) * (NHID / 64);  // 16x64 strips, M=128 -> 6250
    const int waves2 = (NN / 16) * (NOUT / 64);  // M=64 -> 3125

    // ---- GCN layer 1: h1 = x@W1 ; prop ; +b ; relu ----
    gemm_wmma4<<<cdiv(waves1, 8), T, 0, stream>>>(Ap_x, Wp_g1, nullptr, BUF1, NN, NFEAT, NHID);
    fill_kernel<<<cdiv(NN * NHID, T), T, 0, stream>>>(BUF2, 0.0f, NN * NHID);
    scatter_kernel<<<cdiv(EE * (NHID / 8), T), T, 0, stream>>>(BUF1, src, dst, nrm, BUF2, EE, NHID);
    finalize_gcn<<<cdiv(NN * NHID, T), T, 0, stream>>>(BUF2, BUF1, deg, gb1, NN, NHID, 1);

    // ---- GCN layer 2: h2 = x1@W2 ; prop ; +b ----
    pack_a_bf16<<<cdiv(NN * NHID, T), T, 0, stream>>>(BUF2, Ap_h, NN, NHID);
    gemm_wmma4<<<cdiv(waves2, 8), T, 0, stream>>>(Ap_h, Wp_g2, nullptr, SBUF1, NN, NHID, NOUT);
    fill_kernel<<<cdiv(NN * NOUT, T), T, 0, stream>>>(SBUF2, 0.0f, NN * NOUT);
    scatter_kernel<<<cdiv(EE * (NOUT / 8), T), T, 0, stream>>>(SBUF1, src, dst, nrm, SBUF2, EE, NOUT);
    finalize_gcn<<<cdiv(NN * NOUT, T), T, 0, stream>>>(SBUF2, SBUF1, deg, gb2, NN, NOUT, 0);
    // x2 in SBUF2

    // ---- hypergraph degrees ----
    fill_kernel<<<cdiv(NUM_E, T), T, 0, stream>>>(e_deg, 0.0f, NUM_E);
    fill_kernel<<<cdiv(NN, T), T, 0, stream>>>(v_deg, 0.0f, NN);
    count_kernel<<<cdiv(NNZ, T), T, 0, stream>>>(he, e_deg, NNZ);
    count_kernel<<<cdiv(NNZ, T), T, 0, stream>>>(hv, v_deg, NNZ);

    // ---- HGNNP layer 1: h3 = x@hW1 + b ; v->e mean ; e->v mean ; relu ----
    gemm_wmma4<<<cdiv(waves1, 8), T, 0, stream>>>(Ap_x, Wp_h1, hb1, BUF1, NN, NFEAT, NHID);
    fill_kernel<<<cdiv(NUM_E * NHID, T), T, 0, stream>>>(e_acc, 0.0f, NUM_E * NHID);
    scatter_kernel<<<cdiv(NNZ * (NHID / 8), T), T, 0, stream>>>(BUF1, hv, he, nullptr, e_acc, NNZ, NHID);
    div_rows<<<cdiv(NUM_E * NHID, T), T, 0, stream>>>(e_acc, e_deg, NUM_E, NHID, 0);
    fill_kernel<<<cdiv(NN * NHID, T), T, 0, stream>>>(BUF2, 0.0f, NN * NHID);
    scatter_kernel<<<cdiv(NNZ * (NHID / 8), T), T, 0, stream>>>(e_acc, he, hv, nullptr, BUF2, NNZ, NHID);
    div_rows<<<cdiv(NN * NHID, T), T, 0, stream>>>(BUF2, v_deg, NN, NHID, 1);
    // x3 in BUF2

    // ---- HGNNP layer 2: h4 = x3@hW2 + b ; v->e mean ; e->v mean ----
    pack_a_bf16<<<cdiv(NN * NHID, T), T, 0, stream>>>(BUF2, Ap_h, NN, NHID);
    gemm_wmma4<<<cdiv(waves2, 8), T, 0, stream>>>(Ap_h, Wp_h2, hb2, SBUF1, NN, NHID, NOUT);
    fill_kernel<<<cdiv(NUM_E * NOUT, T), T, 0, stream>>>(e_acc, 0.0f, NUM_E * NOUT);
    scatter_kernel<<<cdiv(NNZ * (NOUT / 8), T), T, 0, stream>>>(SBUF1, hv, he, nullptr, e_acc, NNZ, NOUT);
    div_rows<<<cdiv(NUM_E * NOUT, T), T, 0, stream>>>(e_acc, e_deg, NUM_E, NOUT, 0);
    fill_kernel<<<cdiv(NN * NOUT, T), T, 0, stream>>>(BUF1, 0.0f, NN * NOUT);
    scatter_kernel<<<cdiv(NNZ * (NOUT / 8), T), T, 0, stream>>>(e_acc, he, hv, nullptr, BUF1, NNZ, NOUT);
    div_rows<<<cdiv(NN * NOUT, T), T, 0, stream>>>(BUF1, v_deg, NN, NOUT, 0);
    // x4 in BUF1 (first N*NOUT)

    // ---- embeds, row-normalized data ----
    combine_kernel<<<cdiv(NN * NOUT, T), T, 0, stream>>>(SBUF2, BUF1, out_emb, NN * NOUT);
    rownorm_kernel<<<cdiv(NN, T), T, 0, stream>>>(out_emb, data);

    // ---- soft k-means (gated on device-side num_iter) ----
    init_mu_kernel<<<cdiv(KC * NOUT, T), T, 0, stream>>>(data, mu);
    for (int it = 0; it < ITER_MAX; ++it) {
        gated_zero<<<cdiv(KC * NOUT, T), T, 0, stream>>>(cs, KC * NOUT, ni, it);
        gated_zero<<<1, KC, 0, stream>>>(cr, KC, ni, it);
        kmeans_accum<<<cdiv(NN, T), T, 0, stream>>>(data, mu, cr, cs, ni, it);
        mu_update<<<cdiv(KC * NOUT, T), T, 0, stream>>>(mu, cs, cr, ni, it);
    }
    copy_kernel<<<cdiv(KC * NOUT, T), T, 0, stream>>>(mu, out_mu, KC * NOUT);
    kmeans_final<<<cdiv(NN, T), T, 0, stream>>>(data, mu, out_r, out_dist);
}
